// ScalarAttentionModule_82248623719065
// MI455X (gfx1250) — matrix operand order repaired and checked by reading, not compiled, not run on hardware
//
#include <hip/hip_runtime.h>

// ---------------------------------------------------------------------------
// Problem constants (match reference)
// ---------------------------------------------------------------------------
constexpr int Bsz = 8;
constexpr int Nsz = 2048;
constexpr int Fsz = 512;
constexpr int Asz = 512;
constexpr float BN_EPS = 1e-5f;

typedef __attribute__((ext_vector_type(16))) __bf16 v16bf;
typedef __attribute__((ext_vector_type(8)))  float  v8f;
using bf16_t = __bf16;

// LDS staging geometry: every GEMM stages a 16-row x 512-col bf16 A-tile.
constexpr int KC      = 512;          // staged K-chunk
constexpr int LDA_PAD = KC + 8;       // padded row stride (bf16 elems)
constexpr int STAGE_BYTES = 16 * LDA_PAD * 2;   // 16640 B per stage buffer

// ---------------------------------------------------------------------------
// Workspace layout (bytes)
// ---------------------------------------------------------------------------
constexpr size_t OFF_XBF  = 0;                                   // x bf16  [B*N, F]
constexpr size_t OFF_WQB  = OFF_XBF + (size_t)Bsz*Nsz*Fsz*2;
constexpr size_t OFF_WKB  = OFF_WQB + (size_t)Asz*Fsz*2;
constexpr size_t OFF_WVB  = OFF_WKB + (size_t)Asz*Fsz*2;
constexpr size_t OFF_WOB  = OFF_WVB + (size_t)Asz*Fsz*2;
constexpr size_t OFF_Q    = OFF_WOB + (size_t)Fsz*Asz*2;         // Q bf16 [B, N, A]
constexpr size_t OFF_K    = OFF_Q   + (size_t)Bsz*Nsz*Asz*2;
constexpr size_t OFF_VT   = OFF_K   + (size_t)Bsz*Nsz*Asz*2;     // V^T bf16 [B, A, N]
constexpr size_t OFF_FEAT = OFF_VT  + (size_t)Bsz*Asz*Nsz*2;
constexpr size_t OFF_LIN  = OFF_FEAT+ (size_t)Bsz*Nsz*Asz*2;     // lin fp32 [B, N, F]
constexpr size_t OFF_STAT = OFF_LIN + (size_t)Bsz*Nsz*Fsz*4;
constexpr size_t OFF_S    = OFF_STAT+ (size_t)Nsz*2*4;           // S/P bf16 [B, N, N]

// ---------------------------------------------------------------------------
// WMMA + fragment helpers
// ---------------------------------------------------------------------------
__device__ inline v8f bf16_wmma(v16bf a, v16bf b, v8f c) {
  return __builtin_amdgcn_wmma_f32_16x16x32_bf16(
      false, a, false, b, (short)0, c, false, false);
}

union FragU {
  v16bf v;
  uint4 q[2];
};

// B fragment: 32x16 (KxN). Source laid out so each output-column's K-run is
// contiguous with leading dim `ld`. Lane L: col = col0 + (L&15);
// K = k0 + (L<16 ? {0..7,16..23} : {8..15,24..31})  (two 16B global loads).
__device__ inline v16bf load_b_frag(const bf16_t* p, int ld, int col0, int k0, int lane) {
  int col = col0 + (lane & 15);
  int kb  = k0 + ((lane & 16) ? 8 : 0);
  const bf16_t* base = p + (size_t)col * ld + kb;
  FragU u;
  u.q[0] = *(const uint4*)(base);
  u.q[1] = *(const uint4*)(base + 16);
  return u.v;
}

// A fragment from the LDS-staged tile (padded stride LDA_PAD).
// Lane L: row = L&15; K = k0 + {0..7,16..23} (L<16) or {8..15,24..31}.
__device__ inline v16bf lds_a_frag(const bf16_t* tile, int k0, int lane) {
  int row = lane & 15;
  int kb  = k0 + ((lane & 16) ? 8 : 0);
  const bf16_t* p = tile + row * LDA_PAD + kb;
  FragU u;
  u.q[0] = *(const uint4*)(p);        // ds_load_b128
  u.q[1] = *(const uint4*)(p + 16);   // ds_load_b128
  return u.v;
}

__device__ inline unsigned pack2_bf16(float lo, float hi) {
  unsigned short a = __builtin_bit_cast(unsigned short, (bf16_t)lo);
  unsigned short b = __builtin_bit_cast(unsigned short, (bf16_t)hi);
  return (unsigned)a | ((unsigned)b << 16);
}

// ---------------------------------------------------------------------------
// Async global->LDS staging (CDNA5 ASYNCcnt path)
// ---------------------------------------------------------------------------
__device__ inline unsigned lds_off_of(const void* p) {
  // generic -> LDS addrspace -> byte offset relative to wave LDS base
  return (unsigned)(unsigned long long)
      (__attribute__((address_space(3))) const char*)p;
}

__device__ inline void async_g2l_b128(const bf16_t* gsrc, void* ldst) {
  unsigned loff = lds_off_of(ldst);
  asm volatile("global_load_async_to_lds_b128 %0, %1, off"
               :: "v"(loff), "v"(gsrc) : "memory");
}

__device__ inline void wait_async_0() {
#if __has_builtin(__builtin_amdgcn_s_wait_asynccnt)
  __builtin_amdgcn_s_wait_asynccnt(0);
#else
  asm volatile("s_wait_asynccnt 0x0" ::: "memory");
#endif
}
__device__ inline void wait_async_4() {
#if __has_builtin(__builtin_amdgcn_s_wait_asynccnt)
  __builtin_amdgcn_s_wait_asynccnt(4);
#else
  asm volatile("s_wait_asynccnt 0x4" ::: "memory");
#endif
}

// Issue async copy of a 16 x 512 bf16 sub-tile (rows row0..row0+15, cols
// k0..k0+511 of a row-major [*, ld] matrix) into stage buffer `buf`.
// 1024 b128 chunks over 256 threads -> 4 per thread (4 ASYNCcnt per wave).
__device__ inline void stage_issue(const bf16_t* gbase, int ld, int row0, int k0,
                                   char* buf, int tid) {
#pragma unroll
  for (int i = 0; i < 4; ++i) {
    int idx = tid + i * 256;
    int r   = idx >> 6;
    int c8  = (idx & 63) << 3;
    const bf16_t* g = gbase + (size_t)(row0 + r) * ld + (k0 + c8);
    async_g2l_b128(g, buf + ((size_t)r * LDA_PAD + c8) * 2);
  }
}

// ---------------------------------------------------------------------------
// Kernel 0: fp32 -> bf16 conversion (grid-stride)
// ---------------------------------------------------------------------------
__global__ void cvt_bf16_kernel(const float* __restrict__ src,
                                bf16_t* __restrict__ dst, int n) {
  for (int i = blockIdx.x * blockDim.x + threadIdx.x; i < n;
       i += gridDim.x * blockDim.x) {
    dst[i] = (bf16_t)src[i];
  }
}

// ---------------------------------------------------------------------------
// Kernel 1: QKV projection. Block = (which, row-tile); 8 waves cover all 512
// output cols. x row-strip staged in LDS once (K = F = 512, single stage).
// Q/K row-major bf16 [B,N,A]; V stored transposed as V^T [B,A,N].
// ---------------------------------------------------------------------------
__global__ __launch_bounds__(256) void qkv_kernel(
    const bf16_t* __restrict__ xbf,
    const bf16_t* __restrict__ wq, const bf16_t* __restrict__ wk,
    const bf16_t* __restrict__ wv,
    const float* __restrict__ bq, const float* __restrict__ bk,
    const float* __restrict__ bv,
    bf16_t* __restrict__ Q, bf16_t* __restrict__ K, bf16_t* __restrict__ VT) {
  extern __shared__ char smem[];
  const int tid = threadIdx.x, lane = tid & 31, wave = tid >> 5;
  const int which = blockIdx.x / (Bsz * Nsz / 16);   // 0=Q 1=K 2=V
  const int nt    = blockIdx.x % (Bsz * Nsz / 16);
  const int row0  = nt * 16;
  const int a0    = wave * 64;

  const bf16_t* W    = (which == 0) ? wq : (which == 1) ? wk : wv;
  const float*  bias = (which == 0) ? bq : (which == 1) ? bk : bv;

  stage_issue(xbf, Fsz, row0, 0, smem, tid);
  wait_async_0();
  __syncthreads();
  const bf16_t* tile = (const bf16_t*)smem;

  v8f acc[4] = {v8f{}, v8f{}, v8f{}, v8f{}};
  for (int k = 0; k < Fsz; k += 32) {
    v16bf a = lds_a_frag(tile, k, lane);
    if (k + 32 < Fsz)
      __builtin_prefetch(W + (size_t)(a0 + (lane & 15)) * Fsz + k + 32, 0, 3);
#pragma unroll
    for (int j = 0; j < 4; ++j) {
      v16bf b = load_b_frag(W, Fsz, a0 + 16 * j, k, lane);
      acc[j] = bf16_wmma(a, b, acc[j]);
    }
  }

  const int rbase = (lane & 16) ? 8 : 0;
  if (which < 2) {
    bf16_t* Out = (which == 0) ? Q : K;
#pragma unroll
    for (int j = 0; j < 4; ++j) {
      int col = a0 + 16 * j + (lane & 15);
      float bia = bias[col];
#pragma unroll
      for (int r = 0; r < 8; ++r)
        Out[(size_t)(row0 + r + rbase) * Asz + col] = (bf16_t)(acc[j][r] + bia);
    }
  } else {
    int bidx = row0 / Nsz;
    int nloc = row0 % Nsz;
    bf16_t* vt = VT + (size_t)bidx * Asz * Nsz;
#pragma unroll
    for (int j = 0; j < 4; ++j) {
      int col = a0 + 16 * j + (lane & 15);
      float bia = bias[col];
      uint4 pk;
      pk.x = pack2_bf16(acc[j][0] + bia, acc[j][1] + bia);
      pk.y = pack2_bf16(acc[j][2] + bia, acc[j][3] + bia);
      pk.z = pack2_bf16(acc[j][4] + bia, acc[j][5] + bia);
      pk.w = pack2_bf16(acc[j][6] + bia, acc[j][7] + bia);
      *(uint4*)(vt + (size_t)col * Nsz + nloc + rbase) = pk;
    }
  }
}

// ---------------------------------------------------------------------------
// Kernel 2: S = Q @ K^T (bf16 [B,N,N]). Block = (b, row-tile, m-quarter);
// 8 waves cover 512 m-cols. Q row-strip staged in LDS (K = A = 512).
// ---------------------------------------------------------------------------
__global__ __launch_bounds__(256) void scores_kernel(
    const bf16_t* __restrict__ Q, const bf16_t* __restrict__ K,
    bf16_t* __restrict__ S) {
  extern __shared__ char smem[];
  const int tid = threadIdx.x, lane = tid & 31, wave = tid >> 5;
  int b   = blockIdx.x / (128 * 4);
  int rem = blockIdx.x % (128 * 4);
  int nt = rem / 4, mq = rem % 4;
  int row0 = nt * 16;
  int m0   = mq * 512 + wave * 64;

  const bf16_t* Qb = Q + (size_t)b * Nsz * Asz;
  const bf16_t* Kb = K + (size_t)b * Nsz * Asz;
  bf16_t*       Sb = S + (size_t)b * Nsz * Nsz;

  stage_issue(Qb, Asz, row0, 0, smem, tid);
  wait_async_0();
  __syncthreads();
  const bf16_t* tile = (const bf16_t*)smem;

  v8f acc[4] = {v8f{}, v8f{}, v8f{}, v8f{}};
  for (int k = 0; k < Asz; k += 32) {
    v16bf a = lds_a_frag(tile, k, lane);
    if (k + 32 < Asz)
      __builtin_prefetch(Kb + (size_t)(m0 + (lane & 15)) * Asz + k + 32, 0, 3);
#pragma unroll
    for (int j = 0; j < 4; ++j) {
      v16bf bb = load_b_frag(Kb, Asz, m0 + 16 * j, k, lane);
      acc[j] = bf16_wmma(a, bb, acc[j]);
    }
  }

  const int rbase = (lane & 16) ? 8 : 0;
#pragma unroll
  for (int j = 0; j < 4; ++j) {
    int col = m0 + 16 * j + (lane & 15);
#pragma unroll
    for (int r = 0; r < 8; ++r)
      Sb[(size_t)(row0 + r + rbase) * Nsz + col] = (bf16_t)acc[j][r];
  }
}

// ---------------------------------------------------------------------------
// Kernel 3: row softmax over S (in place), one block per (b,n) row.
// ---------------------------------------------------------------------------
__global__ __launch_bounds__(256) void softmax_kernel(bf16_t* __restrict__ S) {
  __shared__ float red[256];
  const int tid = threadIdx.x;
  bf16_t* p = S + (size_t)blockIdx.x * Nsz;

  float v[8];
#pragma unroll
  for (int i = 0; i < 8; ++i) v[i] = (float)p[tid + i * 256];

  float m = v[0];
#pragma unroll
  for (int i = 1; i < 8; ++i) m = fmaxf(m, v[i]);
  red[tid] = m;
  __syncthreads();
  for (int s = 128; s > 0; s >>= 1) {
    if (tid < s) red[tid] = fmaxf(red[tid], red[tid + s]);
    __syncthreads();
  }
  float rowmax = red[0];
  __syncthreads();

  float e[8], sum = 0.f;
#pragma unroll
  for (int i = 0; i < 8; ++i) {
    e[i] = __expf(v[i] - rowmax);
    sum += e[i];
  }
  red[tid] = sum;
  __syncthreads();
  for (int s = 128; s > 0; s >>= 1) {
    if (tid < s) red[tid] += red[tid + s];
    __syncthreads();
  }
  float inv = 1.0f / red[0];
#pragma unroll
  for (int i = 0; i < 8; ++i) p[tid + i * 256] = (bf16_t)(e[i] * inv);
}

// ---------------------------------------------------------------------------
// Kernel 4: feature = P @ V (K = N = 2048). Block = (b, row-tile); 8 waves
// cover all 512 a-cols. P row-strip staged in 4 double-buffered async stages.
// ---------------------------------------------------------------------------
__global__ __launch_bounds__(256) void feature_kernel(
    const bf16_t* __restrict__ P, const bf16_t* __restrict__ VT,
    bf16_t* __restrict__ FEAT) {
  extern __shared__ char smem[];
  const int tid = threadIdx.x, lane = tid & 31, wave = tid >> 5;
  int b  = blockIdx.x / 128;
  int nt = blockIdx.x % 128;
  int row0 = nt * 16;
  int a0   = wave * 64;

  const bf16_t* Pb = P    + (size_t)b * Nsz * Nsz;
  const bf16_t* Vb = VT   + (size_t)b * Asz * Nsz;
  bf16_t*       Fb = FEAT + (size_t)b * Nsz * Asz;

  constexpr int KS = Nsz / KC;   // 4 stages

  stage_issue(Pb, Nsz, row0, 0, smem, tid);

  v8f acc[4] = {v8f{}, v8f{}, v8f{}, v8f{}};
  for (int s = 0; s < KS; ++s) {
    if (s + 1 < KS) {
      stage_issue(Pb, Nsz, row0, (s + 1) * KC,
                  smem + ((s + 1) & 1) * STAGE_BYTES, tid);
      wait_async_4();          // stage s landed; stage s+1 stays in flight
    } else {
      wait_async_0();
    }
    __syncthreads();

    const bf16_t* tile = (const bf16_t*)(smem + (s & 1) * STAGE_BYTES);
    const int kg0 = s * KC;
    for (int k = 0; k < KC; k += 32) {
      v16bf a = lds_a_frag(tile, k, lane);
      __builtin_prefetch(Vb + (size_t)(a0 + (lane & 15)) * Nsz + kg0 + k + 32, 0, 3);
#pragma unroll
      for (int j = 0; j < 4; ++j) {
        v16bf bb = load_b_frag(Vb, Nsz, a0 + 16 * j, kg0 + k, lane);
        acc[j] = bf16_wmma(a, bb, acc[j]);
      }
    }
    __syncthreads();           // protect buffer before stage s+2 overwrite
  }

  const int rbase = (lane & 16) ? 8 : 0;
#pragma unroll
  for (int j = 0; j < 4; ++j) {
    int col = a0 + 16 * j + (lane & 15);
#pragma unroll
    for (int r = 0; r < 8; ++r)
      Fb[(size_t)(row0 + r + rbase) * Asz + col] = (bf16_t)acc[j][r];
  }
}

// ---------------------------------------------------------------------------
// Kernel 5: lin = feature @ Wo^T + bo (fp32 [B*N, F]). Block = row-tile.
// ---------------------------------------------------------------------------
__global__ __launch_bounds__(256) void outproj_kernel(
    const bf16_t* __restrict__ FEAT, const bf16_t* __restrict__ WO,
    const float* __restrict__ bo, float* __restrict__ LIN) {
  extern __shared__ char smem[];
  const int tid = threadIdx.x, lane = tid & 31, wave = tid >> 5;
  int row0 = blockIdx.x * 16;
  int f0   = wave * 64;

  stage_issue(FEAT, Asz, row0, 0, smem, tid);
  wait_async_0();
  __syncthreads();
  const bf16_t* tile = (const bf16_t*)smem;

  v8f acc[4] = {v8f{}, v8f{}, v8f{}, v8f{}};
  for (int k = 0; k < Asz; k += 32) {
    v16bf a = lds_a_frag(tile, k, lane);
#pragma unroll
    for (int j = 0; j < 4; ++j) {
      v16bf bb = load_b_frag(WO, Asz, f0 + 16 * j, k, lane);
      acc[j] = bf16_wmma(a, bb, acc[j]);
    }
  }

  const int rbase = (lane & 16) ? 8 : 0;
#pragma unroll
  for (int j = 0; j < 4; ++j) {
    int col = f0 + 16 * j + (lane & 15);
    float bia = bo[col];
#pragma unroll
    for (int r = 0; r < 8; ++r)
      LIN[(size_t)(row0 + r + rbase) * Fsz + col] = acc[j][r] + bia;
  }
}

// ---------------------------------------------------------------------------
// Kernel 6: BN stats per token channel n over (B, F) = 4096 values.
// ---------------------------------------------------------------------------
__global__ __launch_bounds__(256) void stats_kernel(
    const float* __restrict__ LIN, float* __restrict__ stats) {
  __shared__ float s1[256];
  __shared__ float s2[256];
  const int n = blockIdx.x, tid = threadIdx.x;
  float sum = 0.f, sq = 0.f;
  for (int b = 0; b < Bsz; ++b) {
    const float* p = LIN + ((size_t)b * Nsz + n) * Fsz;
    for (int f = tid; f < Fsz; f += 256) {
      float x = p[f];
      sum += x;
      sq += x * x;
    }
  }
  s1[tid] = sum;
  s2[tid] = sq;
  __syncthreads();
  for (int s = 128; s > 0; s >>= 1) {
    if (tid < s) {
      s1[tid] += s1[tid + s];
      s2[tid] += s2[tid + s];
    }
    __syncthreads();
  }
  if (tid == 0) {
    const float cnt = (float)(Bsz * Fsz);
    float mean = s1[0] / cnt;
    float var  = s2[0] / cnt - mean * mean;
    stats[2 * n]     = mean;
    stats[2 * n + 1] = rsqrtf(var + BN_EPS);
  }
}

// ---------------------------------------------------------------------------
// Kernel 7: out = relu(gamma*(lin-mean)*inv + beta) + x
// ---------------------------------------------------------------------------
__global__ __launch_bounds__(256) void apply_kernel(
    const float* __restrict__ LIN, const float* __restrict__ x,
    const float* __restrict__ stats, const float* __restrict__ gamma,
    const float* __restrict__ beta, float* __restrict__ out) {
  size_t idx = (size_t)blockIdx.x * blockDim.x + threadIdx.x;
  if (idx >= (size_t)Bsz * Nsz * Fsz) return;
  int n = (int)((idx / Fsz) % Nsz);
  float mean = stats[2 * n];
  float inv  = stats[2 * n + 1];
  float v = gamma[n] * (LIN[idx] - mean) * inv + beta[n];
  out[idx] = fmaxf(v, 0.f) + x[idx];
}

// ---------------------------------------------------------------------------
// Host-side launch
// ---------------------------------------------------------------------------
extern "C" void kernel_launch(void* const* d_in, const int* in_sizes, int n_in,
                              void* d_out, int out_size, void* d_ws, size_t ws_size,
                              hipStream_t stream) {
  (void)in_sizes; (void)n_in; (void)out_size; (void)ws_size;
  const float* x     = (const float*)d_in[0];
  const float* Wq    = (const float*)d_in[1];
  const float* bq    = (const float*)d_in[2];
  const float* Wk    = (const float*)d_in[3];
  const float* bk    = (const float*)d_in[4];
  const float* Wv    = (const float*)d_in[5];
  const float* bv    = (const float*)d_in[6];
  const float* Wo    = (const float*)d_in[7];
  const float* bo    = (const float*)d_in[8];
  const float* gamma = (const float*)d_in[9];
  const float* beta  = (const float*)d_in[10];
  float* out = (float*)d_out;

  char* ws = (char*)d_ws;
  bf16_t* XBF  = (bf16_t*)(ws + OFF_XBF);
  bf16_t* WQB  = (bf16_t*)(ws + OFF_WQB);
  bf16_t* WKB  = (bf16_t*)(ws + OFF_WKB);
  bf16_t* WVB  = (bf16_t*)(ws + OFF_WVB);
  bf16_t* WOB  = (bf16_t*)(ws + OFF_WOB);
  bf16_t* Qb   = (bf16_t*)(ws + OFF_Q);
  bf16_t* Kb   = (bf16_t*)(ws + OFF_K);
  bf16_t* VTb  = (bf16_t*)(ws + OFF_VT);
  bf16_t* FEAT = (bf16_t*)(ws + OFF_FEAT);
  float*  LIN  = (float*)(ws + OFF_LIN);
  float*  STAT = (float*)(ws + OFF_STAT);
  bf16_t* Sb   = (bf16_t*)(ws + OFF_S);

  // 0) fp32 -> bf16 conversions
  cvt_bf16_kernel<<<4096, 256, 0, stream>>>(x, XBF, Bsz * Nsz * Fsz);
  cvt_bf16_kernel<<<512, 256, 0, stream>>>(Wq, WQB, Asz * Fsz);
  cvt_bf16_kernel<<<512, 256, 0, stream>>>(Wk, WKB, Asz * Fsz);
  cvt_bf16_kernel<<<512, 256, 0, stream>>>(Wv, WVB, Asz * Fsz);
  cvt_bf16_kernel<<<512, 256, 0, stream>>>(Wo, WOB, Fsz * Asz);

  // 1) QKV projections: 3 * 1024 blocks, async-staged x strip
  qkv_kernel<<<3 * (Bsz * Nsz / 16), 256, STAGE_BYTES, stream>>>(
      XBF, WQB, WKB, WVB, bq, bk, bv, Qb, Kb, VTb);
  // 2) scores: 8 * 128 * 4 blocks, async-staged Q strip
  scores_kernel<<<Bsz * 128 * 4, 256, STAGE_BYTES, stream>>>(Qb, Kb, Sb);
  // 3) softmax: one block per row
  softmax_kernel<<<Bsz * Nsz, 256, 0, stream>>>(Sb);
  // 4) feature = P @ V: 8 * 128 blocks, double-buffered async pipeline
  feature_kernel<<<Bsz * 128, 256, 2 * STAGE_BYTES, stream>>>(Sb, VTb, FEAT);
  // 5) output projection: 1024 blocks
  outproj_kernel<<<Bsz * Nsz / 16, 256, STAGE_BYTES, stream>>>(FEAT, WOB, bo, LIN);
  // 6) BN statistics per channel n
  stats_kernel<<<Nsz, 256, 0, stream>>>(LIN, STAT);
  // 7) normalize + relu + residual
  apply_kernel<<<(Bsz * Nsz * Fsz + 255) / 256, 256, 0, stream>>>(
      LIN, x, STAT, gamma, beta, out);
}
